// MiniBatchGraphSAGE_65996467470993
// MI455X (gfx1250) — matrix operand order repaired and checked by reading, not compiled, not run on hardware
//
#include <hip/hip_runtime.h>

// ---------------------------------------------------------------------------
// MiniBatch GraphSAGE forward for MI455X (gfx1250, wave32, WMMA).
// Activations/weights pre-converted to bf16; GEMMs are pure
// global_load_b128 -> v_wmma_f32_16x16x32_bf16 with f32 accumulate,
// register-blocked 16x64 per wave (A-fragment reuse across 4 B tiles).
// ---------------------------------------------------------------------------

typedef __bf16 v16bf __attribute__((ext_vector_type(16)));
typedef __bf16 v8bf  __attribute__((ext_vector_type(8)));
typedef float  v8f   __attribute__((ext_vector_type(8)));

#define BN_EPS 1e-5f

// float -> bf16, round-to-nearest-even
static __device__ __forceinline__ unsigned short cvt_bf16_bits(float f) {
    unsigned u = __builtin_bit_cast(unsigned, f);
    u += 0x7FFFu + ((u >> 16) & 1u);
    return (unsigned short)(u >> 16);
}

// ---------------------------------------------------------------------------
// utility kernels
// ---------------------------------------------------------------------------
__global__ void k_zero(float* __restrict__ p, long long n) {
    long long i = (long long)blockIdx.x * blockDim.x + threadIdx.x;
    if (i < n) p[i] = 0.0f;
}

// f32 -> bf16 bulk conversion, 4 elements / thread (b128 in, b64 out)
__global__ void k_cvt_bf16(const float* __restrict__ in, unsigned short* __restrict__ out,
                           long long n4) {
    long long i = (long long)blockIdx.x * blockDim.x + threadIdx.x;
    if (i >= n4) return;
    const float4 v = ((const float4*)in)[i];
    ushort4 o;
    o.x = cvt_bf16_bits(v.x);
    o.y = cvt_bf16_bits(v.y);
    o.z = cvt_bf16_bits(v.z);
    o.w = cvt_bf16_bits(v.w);
    ((ushort4*)out)[i] = o;
}

__global__ void k_degree(const int* __restrict__ dst, float* __restrict__ deg, int E) {
    int e = blockIdx.x * blockDim.x + threadIdx.x;
    if (e < E) atomicAdd(&deg[dst[e]], 1.0f);
}

// agg[dst] += feat[src]; float4 granularity; atomics land in L2 (feat L2-resident)
__global__ void k_scatter_add(const float* __restrict__ feat, const int* __restrict__ src,
                              const int* __restrict__ dst, float* __restrict__ agg,
                              int E, int D) {
    const int vecs = D >> 2;
    long long idx = (long long)blockIdx.x * blockDim.x + threadIdx.x;
    if (idx >= (long long)E * vecs) return;
    int e = (int)(idx / vecs);
    int v = (int)(idx % vecs);
    const float4 val = ((const float4*)(feat + (size_t)src[e] * D))[v];
    float* op = agg + (size_t)dst[e] * D + ((size_t)v << 2);
    atomicAdd(op + 0, val.x);
    atomicAdd(op + 1, val.y);
    atomicAdd(op + 2, val.z);
    atomicAdd(op + 3, val.w);
}

__global__ void k_mean(float* __restrict__ agg, const float* __restrict__ deg,
                       long long total, int D) {
    long long idx = (long long)blockIdx.x * blockDim.x + threadIdx.x;
    if (idx >= total) return;
    agg[idx] /= fmaxf(deg[idx / D], 1.0f);
}

// ---------------------------------------------------------------------------
// bf16 fragment loads (contiguous k order, same for A and B; wave32)
// A (16x32): lane holds rows r=lane&15; 16 contiguous K starting at kb+16*kh  (kh=lane>>4)
// B (32x16) = W^T: lane holds 16 contiguous K of weight row tileN+(lane&15),
//             starting at kb+16*kh.
// ---------------------------------------------------------------------------
static __device__ __forceinline__ v16bf load_a_frag(const __bf16* __restrict__ p) {
    v8bf lo = *(const v8bf*)(p);        // K run 0 (16B aligned)
    v8bf hi = *(const v8bf*)(p + 8);    // K run 1
    return __builtin_shufflevector(lo, hi, 0, 1, 2, 3, 4, 5, 6, 7,
                                           8, 9, 10, 11, 12, 13, 14, 15);
}
static __device__ __forceinline__ v16bf load_b_frag(const __bf16* __restrict__ p) {
    return *(const v16bf*)(p);          // 32B aligned
}

// ---------------------------------------------------------------------------
// Fused SAGE GEMM:  H = A1 @ W1^T + A2 @ W2^T + bias     (all bf16 in, f32 out)
// One wave per 16x64 output strip: 4 accumulators, A fragments reused 4x.
// M%16==0, K%32==0, OUT%64==0 -> no guards, EXEC stays all-ones (WMMA req).
// ---------------------------------------------------------------------------
template <int K, int OUT>
__global__ __launch_bounds__(128) void k_sage_gemm(
    const __bf16* __restrict__ A1, const __bf16* __restrict__ A2,
    const __bf16* __restrict__ W1, const __bf16* __restrict__ W2,
    const float* __restrict__ bias, float* __restrict__ H, int M)
{
    const int lane = threadIdx.x & 31;
    const int wave = blockIdx.x * (blockDim.x >> 5) + (threadIdx.x >> 5);
    const int nGroups = OUT / 64;                       // 16x64 strips per row-block
    const int totalWaves = (M / 16) * nGroups;
    if (wave >= totalWaves) return;                     // wave-uniform exit

    const int tileM = (wave / nGroups) * 16;
    const int tileN = (wave % nGroups) * 64;
    const int r  = lane & 15;
    const int kh = lane >> 4;

    const __bf16* a1row = A1 + (size_t)(tileM + r) * K + 16 * kh;
    const __bf16* a2row = A2 + (size_t)(tileM + r) * K + 16 * kh;
    const __bf16* w1row = W1 + (size_t)(tileN + r) * K + 16 * kh;
    const __bf16* w2row = W2 + (size_t)(tileN + r) * K + 16 * kh;

    v8f acc0 = {}, acc1 = {}, acc2 = {}, acc3 = {};
#pragma unroll
    for (int kb = 0; kb < K; kb += 32) {
        if (kb + 32 < K) __builtin_prefetch(a1row + kb + 32, 0, 0);
        const v16bf a1 = load_a_frag(a1row + kb);
        const v16bf a2 = load_a_frag(a2row + kb);
#pragma unroll
        for (int j = 0; j < 4; ++j) {
            const v16bf b1 = load_b_frag(w1row + (size_t)(16 * j) * K + kb);
            const v16bf b2 = load_b_frag(w2row + (size_t)(16 * j) * K + kb);
            v8f a = (j == 0) ? acc0 : (j == 1) ? acc1 : (j == 2) ? acc2 : acc3;
            a = __builtin_amdgcn_wmma_f32_16x16x32_bf16(
                    false, a1, false, b1, (short)0, a, false, false);
            a = __builtin_amdgcn_wmma_f32_16x16x32_bf16(
                    false, a2, false, b2, (short)0, a, false, false);
            if (j == 0) acc0 = a; else if (j == 1) acc1 = a;
            else if (j == 2) acc2 = a; else acc3 = a;
        }
    }

    // C/D layout: VGPR i -> row tileM + 8*kh + i, col = strip base + (lane&15)
    float* outBase = H + (size_t)(tileM + 8 * kh) * OUT + tileN + r;
#pragma unroll
    for (int j = 0; j < 4; ++j) {
        const v8f a = (j == 0) ? acc0 : (j == 1) ? acc1 : (j == 2) ? acc2 : acc3;
        const float bv = bias[tileN + 16 * j + r];
        float* out = outBase + 16 * j;
#pragma unroll
        for (int i = 0; i < 8; ++i) out[(size_t)i * OUT] = a[i] + bv;
    }
}

// ---------------------------------------------------------------------------
// BatchNorm (training-mode batch stats, biased var) + ReLU
// ---------------------------------------------------------------------------
__global__ void k_bn_stats(const float* __restrict__ h, float* __restrict__ sum,
                           float* __restrict__ sumsq, int M, int D) {
    const int col = threadIdx.x;                        // blockDim.x == D
    const int row0 = blockIdx.x * 64;
    const int rmax = (M - row0 < 64) ? (M - row0) : 64;
    float s = 0.0f, q = 0.0f;
    for (int r2 = 0; r2 < rmax; ++r2) {
        float v = h[(size_t)(row0 + r2) * D + col];
        s += v;
        q = fmaf(v, v, q);
    }
    atomicAdd(&sum[col], s);
    atomicAdd(&sumsq[col], q);
}

__global__ void k_bn_relu(float* __restrict__ h, const float* __restrict__ sum,
                          const float* __restrict__ sumsq, const float* __restrict__ w,
                          const float* __restrict__ b, long long total, int D, float invM) {
    long long idx = (long long)blockIdx.x * blockDim.x + threadIdx.x;
    if (idx >= total) return;
    int col = (int)(idx % D);
    float mu  = sum[col] * invM;
    float var = fmaf(sumsq[col], invM, -mu * mu);
    float v = (h[idx] - mu) * rsqrtf(var + BN_EPS) * w[col] + b[col];
    h[idx] = fmaxf(v, 0.0f);
}

// ---------------------------------------------------------------------------
// global mean pool + classifier head
// ---------------------------------------------------------------------------
__global__ void k_pool_count(const int* __restrict__ batch, float* __restrict__ cnt, int Nn) {
    int i = blockIdx.x * blockDim.x + threadIdx.x;
    if (i < Nn) atomicAdd(&cnt[batch[i]], 1.0f);
}

__global__ void k_pool_sum(const float* __restrict__ h, const int* __restrict__ batch,
                           float* __restrict__ esum, int Nn, int D) {
    long long idx = (long long)blockIdx.x * blockDim.x + threadIdx.x;
    if (idx >= (long long)Nn * D) return;
    int row = (int)(idx / D);
    int col = (int)(idx % D);
    atomicAdd(&esum[(size_t)batch[row] * D + col], h[idx]);
}

__global__ void k_pool_final(const float* __restrict__ esum, const float* __restrict__ cnt,
                             float* __restrict__ emb, int total, int D) {
    int idx = blockIdx.x * blockDim.x + threadIdx.x;
    if (idx < total) emb[idx] = esum[idx] / fmaxf(cnt[idx / D], 1.0f);
}

// logits = relu(emb @ cw1^T + cb1) @ cw2^T + cb2 ; one block per graph
__global__ void k_classifier(const float* __restrict__ emb,
                             const float* __restrict__ cw1, const float* __restrict__ cb1,
                             const float* __restrict__ cw2, const float* __restrict__ cb2,
                             float* __restrict__ logits, int D, int C) {
    __shared__ float hid[256];
    const int g = blockIdx.x;
    const int t = threadIdx.x;                          // blockDim.x == D
    float acc = cb1[t];
    const float* erow = emb + (size_t)g * D;
    const float* wrow = cw1 + (size_t)t * D;
    for (int k = 0; k < D; ++k) acc = fmaf(erow[k], wrow[k], acc);
    hid[t] = fmaxf(acc, 0.0f);
    __syncthreads();
    if (t < C) {
        float a = cb2[t];
        const float* w2row = cw2 + (size_t)t * D;
        for (int k = 0; k < D; ++k) a = fmaf(hid[k], w2row[k], a);
        logits[(size_t)g * C + t] = a;
    }
}

// ---------------------------------------------------------------------------
static inline int gridFor(long long n, int bs) { return (int)((n + bs - 1) / bs); }

extern "C" void kernel_launch(void* const* d_in, const int* in_sizes, int n_in,
                              void* d_out, int out_size, void* d_ws, size_t ws_size,
                              hipStream_t stream)
{
    const float* x    = (const float*)d_in[0];
    const int*   ei   = (const int*)  d_in[1];
    const int*   bat  = (const int*)  d_in[2];
    const float* wl0  = (const float*)d_in[3];
    const float* bl0  = (const float*)d_in[4];
    const float* wr0  = (const float*)d_in[5];
    const float* bnw0 = (const float*)d_in[6];
    const float* bnb0 = (const float*)d_in[7];
    const float* wl1  = (const float*)d_in[8];
    const float* bl1  = (const float*)d_in[9];
    const float* wr1  = (const float*)d_in[10];
    const float* bnw1 = (const float*)d_in[11];
    const float* bnb1 = (const float*)d_in[12];
    const float* cw1  = (const float*)d_in[13];
    const float* cb1  = (const float*)d_in[14];
    const float* cw2  = (const float*)d_in[15];
    const float* cb2  = (const float*)d_in[16];

    const int Nn  = in_sizes[2];             // 50000
    const int E   = in_sizes[1] / 2;         // 800000
    const int DIN = in_sizes[0] / Nn;        // 128
    const int H1  = in_sizes[4];             // 256
    const int H2  = in_sizes[9];             // 128
    const int C   = in_sizes[16];            // 2
    const int Gn  = out_size / (C + H2);     // 64
    const int* src = ei;
    const int* dst = ei + E;

    // ---- workspace carve-up -------------------------------------------------
    // f32: agg | h0 | h1 | deg | csum | csq | cnt | esum
    float* agg  = (float*)d_ws;                     // Nn*H1 (layer0 uses Nn*DIN)
    float* h0   = agg  + (size_t)Nn * H1;           // Nn*H1
    float* h1   = h0   + (size_t)Nn * H1;           // Nn*H2
    float* deg  = h1   + (size_t)Nn * H2;           // Nn
    float* csum = deg  + Nn;                        // H1
    float* csq  = csum + H1;                        // H1
    float* cnt  = csq  + H1;                        // Gn
    float* esum = cnt  + Gn;                        // Gn*H2
    // bf16: aggB | srcB | 4 weight buffers
    __bf16* aggB = (__bf16*)(esum + (size_t)Gn * H2);   // Nn*H1
    __bf16* srcB = aggB + (size_t)Nn * H1;              // Nn*H1 (x or h0 in bf16)
    __bf16* wl0B = srcB + (size_t)Nn * H1;              // H1*DIN
    __bf16* wr0B = wl0B + (size_t)H1 * DIN;             // H1*DIN
    __bf16* wl1B = wr0B + (size_t)H1 * DIN;             // H2*H1
    __bf16* wr1B = wl1B + (size_t)H2 * H1;              // H2*H1

    float* logits = (float*)d_out;                  // [Gn, C]
    float* emb    = logits + (size_t)Gn * C;        // [Gn, H2]

    const int BS = 256;

    // ---- one-time weight conversion to bf16 (tiny, L2-resident) -------------
    k_cvt_bf16<<<gridFor((long long)H1 * DIN / 4, BS), BS, 0, stream>>>(wl0, (unsigned short*)wl0B, (long long)H1 * DIN / 4);
    k_cvt_bf16<<<gridFor((long long)H1 * DIN / 4, BS), BS, 0, stream>>>(wr0, (unsigned short*)wr0B, (long long)H1 * DIN / 4);
    k_cvt_bf16<<<gridFor((long long)H2 * H1  / 4, BS), BS, 0, stream>>>(wl1, (unsigned short*)wl1B, (long long)H2 * H1 / 4);
    k_cvt_bf16<<<gridFor((long long)H2 * H1  / 4, BS), BS, 0, stream>>>(wr1, (unsigned short*)wr1B, (long long)H2 * H1 / 4);

    // ---------------- layer 0: SAGEConv(x) -> h0[N,H1], BN, ReLU -------------
    k_zero<<<gridFor(Nn, BS), BS, 0, stream>>>(deg, Nn);
    k_zero<<<gridFor((long long)Nn * DIN, BS), BS, 0, stream>>>(agg, (long long)Nn * DIN);
    k_degree<<<gridFor(E, BS), BS, 0, stream>>>(dst, deg, E);
    k_scatter_add<<<gridFor((long long)E * (DIN / 4), BS), BS, 0, stream>>>(x, src, dst, agg, E, DIN);
    k_mean<<<gridFor((long long)Nn * DIN, BS), BS, 0, stream>>>(agg, deg, (long long)Nn * DIN, DIN);

    k_cvt_bf16<<<gridFor((long long)Nn * DIN / 4, BS), BS, 0, stream>>>(agg, (unsigned short*)aggB, (long long)Nn * DIN / 4);
    k_cvt_bf16<<<gridFor((long long)Nn * DIN / 4, BS), BS, 0, stream>>>(x,   (unsigned short*)srcB, (long long)Nn * DIN / 4);

    {
        int waves  = (Nn / 16) * (H1 / 64);         // 16x64 strip per wave
        int blocks = (waves + 3) / 4;               // 4 waves / 128-thread block
        k_sage_gemm<128, 256><<<blocks, 128, 0, stream>>>(aggB, srcB, wl0B, wr0B, bl0, h0, Nn);
    }

    k_zero<<<gridFor(2 * H1, BS), BS, 0, stream>>>(csum, 2 * H1);
    k_bn_stats<<<gridFor(Nn, 64), H1, 0, stream>>>(h0, csum, csq, Nn, H1);
    k_bn_relu<<<gridFor((long long)Nn * H1, BS), BS, 0, stream>>>(
        h0, csum, csq, bnw0, bnb0, (long long)Nn * H1, H1, 1.0f / (float)Nn);

    // ---------------- layer 1: SAGEConv(h0) -> h1[N,H2], BN, ReLU ------------
    k_zero<<<gridFor((long long)Nn * H1, BS), BS, 0, stream>>>(agg, (long long)Nn * H1);
    k_scatter_add<<<gridFor((long long)E * (H1 / 4), BS), BS, 0, stream>>>(h0, src, dst, agg, E, H1);
    k_mean<<<gridFor((long long)Nn * H1, BS), BS, 0, stream>>>(agg, deg, (long long)Nn * H1, H1);

    k_cvt_bf16<<<gridFor((long long)Nn * H1 / 4, BS), BS, 0, stream>>>(agg, (unsigned short*)aggB, (long long)Nn * H1 / 4);
    k_cvt_bf16<<<gridFor((long long)Nn * H1 / 4, BS), BS, 0, stream>>>(h0,  (unsigned short*)srcB, (long long)Nn * H1 / 4);

    {
        int waves  = (Nn / 16) * (H2 / 64);
        int blocks = (waves + 3) / 4;
        k_sage_gemm<256, 128><<<blocks, 128, 0, stream>>>(aggB, srcB, wl1B, wr1B, bl1, h1, Nn);
    }

    k_zero<<<gridFor(2 * H1, BS), BS, 0, stream>>>(csum, 2 * H1);
    k_bn_stats<<<gridFor(Nn, 64), H2, 0, stream>>>(h1, csum, csq, Nn, H2);
    k_bn_relu<<<gridFor((long long)Nn * H2, BS), BS, 0, stream>>>(
        h1, csum, csq, bnw1, bnb1, (long long)Nn * H2, H2, 1.0f / (float)Nn);

    // ---------------- global mean pool + classifier head ---------------------
    k_zero<<<gridFor(Gn + (long long)Gn * H2, BS), BS, 0, stream>>>(cnt, Gn + (long long)Gn * H2);
    k_pool_count<<<gridFor(Nn, BS), BS, 0, stream>>>(bat, cnt, Nn);
    k_pool_sum<<<gridFor((long long)Nn * H2, BS), BS, 0, stream>>>(h1, bat, esum, Nn, H2);
    k_pool_final<<<gridFor((long long)Gn * H2, BS), BS, 0, stream>>>(esum, cnt, emb, Gn * H2, H2);
    k_classifier<<<Gn, H2, 0, stream>>>(emb, cw1, cb1, cw2, cb2, logits, H2, C);
}
